// HGTImageFeatureExtractor_41944650613218
// MI455X (gfx1250) — matrix operand order repaired and checked by reading, not compiled, not run on hardware
//
#include <hip/hip_runtime.h>
#include <hip/hip_bf16.h>
#include <limits>

// ---------------------------------------------------------------------------
// Types / feature detection for CDNA5 (gfx1250, wave32)
// ---------------------------------------------------------------------------
typedef __attribute__((ext_vector_type(16))) __bf16 v16bf;
typedef __attribute__((ext_vector_type(8)))  float  v8f;

#define WAVE 32
#define GLOBAL_AS __attribute__((address_space(1)))
#define LDS_AS    __attribute__((address_space(3)))

#if defined(__has_builtin)
#  if __has_builtin(__builtin_amdgcn_global_load_async_to_lds_b32)
#    define HAVE_ASYNC_LDS 1
#  else
#    define HAVE_ASYNC_LDS 0
#  endif
#else
#  define HAVE_ASYNC_LDS 0
#endif

#if HAVE_ASYNC_LDS
__device__ inline void wait_async0() {
#  if __has_builtin(__builtin_amdgcn_s_wait_asynccnt)
    __builtin_amdgcn_s_wait_asynccnt(0);
#  else
    asm volatile("s_wait_asynccnt 0" ::: "memory");
#  endif
}
#endif

// ---------------------------------------------------------------------------
// GEMM:  C[M,N] = act(A[M,K] @ W[K,N] + bias)    (fp32 in/out)
// bf16x3 split accumulation on v_wmma_f32_16x16x32_bf16 (~fp32 accuracy).
// Block = 128 threads = 4 waves; block tile 16(M) x 64(N); per-wave 16x16.
// A (16x32) and B (32x64) tiles staged in LDS via async LDS-DMA when the
// toolchain exposes it (GLOBAL_LOAD_ASYNC_TO_LDS_B32 + s_wait_asynccnt).
// grid = (M/16, N/64). M%16==0, N%64==0, K%32==0.
// ---------------------------------------------------------------------------
template<int ACT>
__global__ void __launch_bounds__(128)
gemm_bias(const float* __restrict__ A, const float* __restrict__ W,
          const float* __restrict__ bias, float* __restrict__ C,
          int M, int K, int N)
{
    __shared__ float As[16 * 32];
    __shared__ float Bs[32 * 64];

    const int tid    = threadIdx.x;
    const int lane   = tid & 31;
    const int wave   = tid >> 5;
    const int m0     = blockIdx.x * 16;
    const int nB     = blockIdx.y * 64;
    const int n0     = nB + wave * 16;
    const int rc     = lane & 15;        // A row within tile == B column within wave tile
    const int hiHalf = lane >> 4;        // lane group 0/1

    v8f acc = {0.f,0.f,0.f,0.f,0.f,0.f,0.f,0.f};

    for (int kk = 0; kk < K; kk += 32) {
        __syncthreads();
#if HAVE_ASYNC_LDS
        for (int i = tid; i < 512; i += 128) {          // A tile: 16x32, coalesced rows
            int r = i >> 5, c = i & 31;
            __builtin_amdgcn_global_load_async_to_lds_b32(
                (GLOBAL_AS int*)(A + (size_t)(m0 + r) * K + kk + c),
                (LDS_AS int*)&As[i], 0, 0);
        }
        for (int i = tid; i < 2048; i += 128) {         // B tile: 32x64, coalesced rows
            int r = i >> 6, c = i & 63;
            __builtin_amdgcn_global_load_async_to_lds_b32(
                (GLOBAL_AS int*)(W + (size_t)(kk + r) * N + nB + c),
                (LDS_AS int*)&Bs[i], 0, 0);
        }
        wait_async0();
#else
        for (int i = tid; i < 512; i += 128) {
            int r = i >> 5, c = i & 31;
            As[i] = A[(size_t)(m0 + r) * K + kk + c];
        }
        for (int i = tid; i < 2048; i += 128) {
            int r = i >> 6, c = i & 63;
            Bs[i] = W[(size_t)(kk + r) * N + nB + c];
        }
#endif
        if (kk + 32 < K) {   // hint next tiles into cache (global_prefetch_b8)
            __builtin_prefetch(&A[(size_t)(m0 + (tid >> 5)) * K + kk + 32 + (tid & 31)], 0, 1);
            __builtin_prefetch(&W[(size_t)(kk + 32 + (tid >> 6)) * N + nB + (tid & 63)], 0, 1);
        }
        __syncthreads();

        v16bf ah, al, bh, bl;
#pragma unroll
        for (int e = 0; e < 16; ++e) {
            // A-matrix 16x32 bf16 lane layout (ISA 7.12.2)
            int ka = ((e & 8) ? 16 : 0) + (hiHalf ? 8 : 0) + (e & 7);
            float av = As[rc * 32 + ka];
            __bf16 avh = (__bf16)av;
            ah[e] = avh;
            al[e] = (__bf16)(av - (float)avh);
            // B-matrix 32x16 bf16: same k order as A (table order)
            int kb = ka;
            float bv = Bs[kb * 64 + wave * 16 + rc];
            __bf16 bvh = (__bf16)bv;
            bh[e] = bvh;
            bl[e] = (__bf16)(bv - (float)bvh);
        }
        // bf16x3: hi*hi + hi*lo + lo*hi  (error ~2^-16 relative)
        acc = __builtin_amdgcn_wmma_f32_16x16x32_bf16(false, ah, false, bh, (short)0, acc, false, false);
        acc = __builtin_amdgcn_wmma_f32_16x16x32_bf16(false, ah, false, bl, (short)0, acc, false, false);
        acc = __builtin_amdgcn_wmma_f32_16x16x32_bf16(false, al, false, bh, (short)0, acc, false, false);
    }

    const int n  = n0 + rc;
    const float bz = bias ? bias[n] : 0.0f;
#pragma unroll
    for (int r = 0; r < 8; ++r) {
        int m = m0 + r + hiHalf * 8;           // C/D layout (ISA 7.12.2)
        float v = acc[r] + bz;
        if (ACT == 1) v = fmaxf(v, 0.0f);
        C[(size_t)m * N + n] = v;
    }
}

// ---------------------------------------------------------------------------
// Rowwise helpers (C = 128, one wave per row, 8 rows per 256-thread block)
// ---------------------------------------------------------------------------
__device__ inline float wave_sum(float v) {
#pragma unroll
    for (int o = 16; o > 0; o >>= 1) v += __shfl_xor(v, o, WAVE);
    return v;
}

// out = LayerNorm(x + y) * g + b     (out may alias x)
__global__ void __launch_bounds__(256)
ln_add(const float* __restrict__ X, const float* __restrict__ Y,
       const float* __restrict__ g, const float* __restrict__ b,
       float* __restrict__ Out, int rows)
{
    int lane = threadIdx.x & 31;
    int row  = blockIdx.x * 8 + (threadIdx.x >> 5);
    if (row >= rows) return;
    const float* x = X + (size_t)row * 128;
    const float* y = Y + (size_t)row * 128;
    float v[4]; float s = 0.f;
#pragma unroll
    for (int i = 0; i < 4; ++i) { v[i] = x[lane + 32*i] + y[lane + 32*i]; s += v[i]; }
    float mu = wave_sum(s) * (1.0f/128.0f);
    float q = 0.f;
#pragma unroll
    for (int i = 0; i < 4; ++i) { float d = v[i] - mu; q += d*d; }
    float inv = rsqrtf(wave_sum(q) * (1.0f/128.0f) + 1e-5f);
#pragma unroll
    for (int i = 0; i < 4; ++i) {
        int c = lane + 32*i;
        Out[(size_t)row*128 + c] = (v[i] - mu) * inv * g[c] + b[c];
    }
}

// h += relu(LN(sig(skip)*o + (1-sig(skip))*h) * g + b)
__global__ void __launch_bounds__(256)
hgt_combine(float* __restrict__ H, const float* __restrict__ O,
            const float* __restrict__ skip,
            const float* __restrict__ g, const float* __restrict__ b, int rows)
{
    int lane = threadIdx.x & 31;
    int row  = blockIdx.x * 8 + (threadIdx.x >> 5);
    if (row >= rows) return;
    float gs = 1.0f / (1.0f + expf(-skip[0]));
    float* h = H + (size_t)row * 128;
    const float* o = O + (size_t)row * 128;
    float hv[4], t[4]; float s = 0.f;
#pragma unroll
    for (int i = 0; i < 4; ++i) {
        int c = lane + 32*i;
        hv[i] = h[c];
        t[i]  = gs * o[c] + (1.0f - gs) * hv[i];
        s += t[i];
    }
    float mu = wave_sum(s) * (1.0f/128.0f);
    float q = 0.f;
#pragma unroll
    for (int i = 0; i < 4; ++i) { float d = t[i] - mu; q += d*d; }
    float inv = rsqrtf(wave_sum(q) * (1.0f/128.0f) + 1e-5f);
#pragma unroll
    for (int i = 0; i < 4; ++i) {
        int c = lane + 32*i;
        float ln = (t[i] - mu) * inv * g[c] + b[c];
        h[c] = hv[i] + fmaxf(ln, 0.0f);
    }
}

__global__ void gelu_ew(const float* __restrict__ X, float* __restrict__ Y, int n) {
    int i = blockIdx.x * 256 + threadIdx.x;
    if (i < n) { float x = X[i]; Y[i] = 0.5f * x * (1.0f + erff(x * 0.70710678118654752f)); }
}

__global__ void fill_f32(float* __restrict__ p, float v, int n) {
    int i = blockIdx.x * 256 + threadIdx.x;
    if (i < n) p[i] = v;
}

__global__ void copy_f32(const float* __restrict__ s, float* __restrict__ d, int n) {
    int i = blockIdx.x * 256 + threadIdx.x;
    if (i < n) d[i] = s[i];
}

// ---------------------------------------------------------------------------
// HGT edge passes.  One wave per edge, lane = feature dim within a head.
// Relation matrices (16 KB) staged in LDS once per block.
// Ordered-int trick for float atomic max (init -inf).
// ---------------------------------------------------------------------------
__device__ inline void atomicMaxF(float* addr, float v) {
    if (v >= 0.0f) atomicMax((int*)addr, __float_as_int(v));
    else           atomicMin((unsigned int*)addr, __float_as_uint(v));
}

__global__ void __launch_bounds__(256)
edge_logit(const int* __restrict__ src, const int* __restrict__ dst, int E,
           const float* __restrict__ Ksrc, const float* __restrict__ Qdst,
           const float* __restrict__ relA, const float* __restrict__ relP,
           float* __restrict__ logit, float* __restrict__ redmax)
{
    __shared__ float aS[4096];   // [H=4][32][32]
    __shared__ float pS[4];
    for (int i = threadIdx.x; i < 4096; i += 256) aS[i] = relA[i];
    if (threadIdx.x < 4) pS[threadIdx.x] = relP[threadIdx.x];
    __syncthreads();

    int lane = threadIdx.x & 31;
    int e = blockIdx.x * 8 + (threadIdx.x >> 5);
    if (e >= E) return;
    int s = src[e], t = dst[e];
    const float inv_sqrt_d = 0.17677669529663687f;  // 1/sqrt(32)
#pragma unroll
    for (int h = 0; h < 4; ++h) {
        float kv = Ksrc[(size_t)s * 128 + h * 32 + lane];
        const float* a = &aS[h * 1024];             // [32][32]
        float kr = 0.f;                             // k_rel[f=lane]
#pragma unroll 8
        for (int d = 0; d < 32; ++d)
            kr = fmaf(__shfl(kv, d, WAVE), a[d * 32 + lane], kr);
        float p = Qdst[(size_t)t * 128 + h * 32 + lane] * kr;
        float lg = wave_sum(p) * pS[h] * inv_sqrt_d;
        if (lane == 0) {
            logit[(size_t)e * 4 + h] = lg;
            atomicMaxF(&redmax[(size_t)t * 4 + h], lg);
        }
    }
}

__global__ void __launch_bounds__(256)
edge_expsum(const int* __restrict__ dst, int E, float* __restrict__ logit,
            const float* __restrict__ redmax, float* __restrict__ redsum)
{
    int i = blockIdx.x * 256 + threadIdx.x;
    if (i >= E * 4) return;
    int e = i >> 2, h = i & 3;
    int t = dst[e];
    float a = expf(logit[i] - redmax[(size_t)t * 4 + h]);
    logit[i] = a;
    atomicAdd(&redsum[(size_t)t * 4 + h], a);
}

__global__ void __launch_bounds__(256)
edge_agg(const int* __restrict__ src, const int* __restrict__ dst, int E,
         const float* __restrict__ Vsrc, const float* __restrict__ relM,
         const float* __restrict__ logit, const float* __restrict__ redsum,
         float* __restrict__ agg)
{
    __shared__ float mS[4096];   // [H=4][32][32]
    for (int i = threadIdx.x; i < 4096; i += 256) mS[i] = relM[i];
    __syncthreads();

    int lane = threadIdx.x & 31;
    int e = blockIdx.x * 8 + (threadIdx.x >> 5);
    if (e >= E) return;
    int s = src[e], t = dst[e];
#pragma unroll
    for (int h = 0; h < 4; ++h) {
        float attn = logit[(size_t)e * 4 + h] / redsum[(size_t)t * 4 + h];
        float vv = Vsrc[(size_t)s * 128 + h * 32 + lane];
        const float* m = &mS[h * 1024];
        float vr = 0.f;
#pragma unroll 8
        for (int d = 0; d < 32; ++d)
            vr = fmaf(__shfl(vv, d, WAVE), m[d * 32 + lane], vr);
        atomicAdd(&agg[(size_t)t * 128 + h * 32 + lane], vr * attn);
    }
}

// ---------------------------------------------------------------------------
// TopK pooling
// ---------------------------------------------------------------------------
__global__ void wnorm_k(const float* __restrict__ w, float* __restrict__ out) {
    __shared__ float part[4];
    int lane = threadIdx.x & 31, wv = threadIdx.x >> 5;
    float v = w[threadIdx.x];
    float s = wave_sum(v * v);
    if (lane == 0) part[wv] = s;
    __syncthreads();
    if (threadIdx.x == 0) out[0] = 1.0f / sqrtf(part[0] + part[1] + part[2] + part[3]);
}

__global__ void __launch_bounds__(256)
score_k(const float* __restrict__ Hp, const float* __restrict__ w,
        const float* __restrict__ invn, float* __restrict__ score, int Np)
{
    int lane = threadIdx.x & 31;
    int n = blockIdx.x * 8 + (threadIdx.x >> 5);
    if (n >= Np) return;
    float s = 0.f;
#pragma unroll
    for (int i = 0; i < 4; ++i)
        s += Hp[(size_t)n * 128 + lane + 32*i] * w[lane + 32*i];
    s = wave_sum(s);
    if (lane == 0) score[n] = tanhf(s * invn[0]);
}

// One block per graph: bitonic-sort 512 scores descending, pool top 256.
__global__ void __launch_bounds__(256)
topk_pool(const float* __restrict__ Hp, const float* __restrict__ score,
          float* __restrict__ out)
{
    __shared__ float s[512];
    __shared__ int   id[512];
    int g = blockIdx.x, tid = threadIdx.x;
    for (int i = tid; i < 512; i += 256) { s[i] = score[g * 512 + i]; id[i] = i; }
    __syncthreads();
    for (int k = 2; k <= 512; k <<= 1) {
        for (int j = k >> 1; j > 0; j >>= 1) {
#pragma unroll
            for (int half = 0; half < 2; ++half) {
                int i = tid + half * 256;
                int ixj = i ^ j;
                if (ixj > i) {
                    bool desc = ((i & k) == 0);
                    float si = s[i], sj = s[ixj];
                    if (desc ? (si < sj) : (si > sj)) {
                        s[i] = sj; s[ixj] = si;
                        int ti = id[i]; id[i] = id[ixj]; id[ixj] = ti;
                    }
                }
            }
            __syncthreads();
        }
    }
    if (tid < 128) {
        float acc = 0.f;
        for (int j = 0; j < 256; ++j)
            acc += s[j] * Hp[((size_t)g * 512 + id[j]) * 128 + tid];
        out[(size_t)g * 128 + tid] = acc;
    }
}

// ---------------------------------------------------------------------------
// Host orchestration
// ---------------------------------------------------------------------------
static void build_leaf_sizes(int* s) {
    int n = 0;
    s[n++] = 512*128; s[n++] = 128; s[n++] = 768*128; s[n++] = 128;     // in_proj
    for (int t = 0; t < 2; ++t) {                                       // tenc x2
        s[n++]=128*128; s[n++]=128; s[n++]=128*128; s[n++]=128;
        s[n++]=128*512; s[n++]=512; s[n++]=512*128; s[n++]=128;
        s[n++]=128; s[n++]=128; s[n++]=128; s[n++]=128;
    }
    for (int l = 0; l < 2; ++l) {                                       // hgt x2
        for (int m = 0; m < 3; ++m) { s[n++]=128*128; s[n++]=128; s[n++]=128*128; s[n++]=128; }
        for (int r = 0; r < 3; ++r) { s[n++]=4096; s[n++]=4096; s[n++]=4; }
        s[n++]=128*128; s[n++]=128; s[n++]=128*128; s[n++]=128;         // a_lin
        s[n++]=1; s[n++]=1;                                             // skip
        s[n++]=128; s[n++]=128; s[n++]=128; s[n++]=128;                 // ln
    }
    s[n++] = 128;                                                       // pool_w
}

extern "C" void kernel_launch(void* const* d_in, const int* in_sizes, int n_in,
                              void* d_out, int out_size, void* d_ws, size_t ws_size,
                              hipStream_t stream)
{
    (void)in_sizes; (void)out_size; (void)ws_size;

    const int NPn = 32768, NTn = 2048, Bg = 64;
    const int E_INTRA = 524288, E_X = 65536;

    const float* xP = (const float*)d_in[0];
    const float* xT = (const float*)d_in[1];

    static int leaf_sz[91];
    build_leaf_sizes(leaf_sz);
    const float* P[91];
    int edgeBase;
    if (n_in >= 93) {                       // one d_in entry per pytree leaf
        for (int i = 0; i < 91; ++i) P[i] = (const float*)d_in[2 + i];
        edgeBase = 93;
    } else {                                // params concatenated in d_in[2]
        const float* base = (const float*)d_in[2];
        size_t off = 0;
        for (int i = 0; i < 91; ++i) { P[i] = base + off; off += leaf_sz[i]; }
        edgeBase = 3;
    }
    const int* eIntra = (const int*)d_in[edgeBase + 0];
    const int* eP2T   = (const int*)d_in[edgeBase + 1];
    const int* eT2P   = (const int*)d_in[edgeBase + 2];

    // workspace layout (floats) — total ~135 MB, fits the 192 MB L2
    float* ws = (float*)d_ws;
    size_t o = 0;
    auto ALLOC = [&](size_t n){ size_t r = o; o += n; return r; };
    const size_t MszP = (size_t)NPn * 128, MszT = (size_t)NTn * 128;
    float* HP  = ws + ALLOC(MszP);
    float* HT  = ws + ALLOC(MszT);
    float* BIG = ws + ALLOC((size_t)NPn * 512);   // FF temp, then Kp/Qp/Vp
    float* TP  = ws + ALLOC(MszP);
    float* TT  = ws + ALLOC((size_t)NTn * 512);   // FF temp, then Kt/Qt/Vt
    float* TT2 = ws + ALLOC(MszT);
    float* AGP = ws + ALLOC(MszP);
    float* AGT = ws + ALLOC(MszT);
    float* LGT = ws + ALLOC((size_t)E_INTRA * 4);
    float* RMX = ws + ALLOC((size_t)NPn * 4);
    float* RSM = ws + ALLOC((size_t)NPn * 4);
    float* SCO = ws + ALLOC((size_t)NPn);
    float* SC1 = ws + ALLOC(16);

    const float NEG_INF = -std::numeric_limits<float>::infinity();

    auto gemm = [&](const float* A, const float* W, const float* b, float* C,
                    int M, int K, int N, int act) {
        dim3 g(M / 16, N / 64);
        if (act) gemm_bias<1><<<g, 128, 0, stream>>>(A, W, b, C, M, K, N);
        else     gemm_bias<0><<<g, 128, 0, stream>>>(A, W, b, C, M, K, N);
    };
    auto fill = [&](float* p, float v, int n) {
        fill_f32<<<(n + 255) / 256, 256, 0, stream>>>(p, v, n);
    };

    // ---- input projections -------------------------------------------------
    gemm(xP, P[0], P[1], HP, NPn, 512, 128, 0);
    gemm(xT, P[2], P[3], HT, NTn, 768, 128, 0);

    // ---- transformer encoder (seq_len==1 => attn == Wo(Wv x)) -------------
    auto tenc = [&](float* Hb, int rows, const float* const* tp,
                    float* t1, float* t2, float* ff) {
        gemm(Hb, tp[0], tp[1], t1, rows, 128, 128, 0);                  // Wv
        gemm(t1, tp[2], tp[3], t2, rows, 128, 128, 0);                  // Wo
        ln_add<<<(rows + 7) / 8, 256, 0, stream>>>(Hb, t2, tp[8], tp[9], Hb, rows);
        gemm(Hb, tp[4], tp[5], ff, rows, 128, 512, 1);                  // W1 + relu
        gemm(ff, tp[6], tp[7], t1, rows, 512, 128, 0);                  // W2
        ln_add<<<(rows + 7) / 8, 256, 0, stream>>>(Hb, t1, tp[10], tp[11], Hb, rows);
    };
    tenc(HP, NPn, &P[4],  TP,  AGP, BIG);
    tenc(HT, NTn, &P[16], TT2, AGT, TT);

    // ---- HGT layers --------------------------------------------------------
    for (int l = 0; l < 2; ++l) {
        const float* const* L = &P[28 + l * 31];
        float *Kp = BIG, *Qp = BIG + MszP, *Vp = BIG + 2 * MszP;
        float *Kt = TT,  *Qt = TT  + MszT, *Vt = TT  + 2 * MszT;
        gemm(HP, L[0],  L[1],  Kp, NPn, 128, 128, 0);
        gemm(HP, L[4],  L[5],  Qp, NPn, 128, 128, 0);
        gemm(HP, L[8],  L[9],  Vp, NPn, 128, 128, 0);
        gemm(HT, L[2],  L[3],  Kt, NTn, 128, 128, 0);
        gemm(HT, L[6],  L[7],  Qt, NTn, 128, 128, 0);
        gemm(HT, L[10], L[11], Vt, NTn, 128, 128, 0);
        fill(AGP, 0.f, (int)MszP);
        fill(AGT, 0.f, (int)MszT);

        struct Rel { const int* e; int E; const float* Ks; const float* Qd;
                     const float* Vs; float* agg; int nd; int ri; };
        Rel rels[3] = {
            { eIntra, E_INTRA, Kp, Qp, Vp, AGP, NPn, 12 },   // intra: p->p
            { eP2T,   E_X,     Kp, Qt, Vp, AGT, NTn, 15 },   // p2t:   p->t
            { eT2P,   E_X,     Kt, Qp, Vt, AGP, NPn, 18 },   // t2p:   t->p
        };
        for (int r = 0; r < 3; ++r) {
            const Rel& R = rels[r];
            fill(RMX, NEG_INF, R.nd * 4);
            fill(RSM, 0.f,     R.nd * 4);
            const int* src = R.e;
            const int* dst = R.e + R.E;
            edge_logit<<<(R.E + 7) / 8, 256, 0, stream>>>(
                src, dst, R.E, R.Ks, R.Qd, L[R.ri], L[R.ri + 2], LGT, RMX);
            edge_expsum<<<(R.E * 4 + 255) / 256, 256, 0, stream>>>(
                dst, R.E, LGT, RMX, RSM);
            edge_agg<<<(R.E + 7) / 8, 256, 0, stream>>>(
                src, dst, R.E, R.Vs, L[R.ri + 1], LGT, RSM, R.agg);
        }

        gelu_ew<<<((int)MszP + 255) / 256, 256, 0, stream>>>(AGP, TP, (int)MszP);
        gemm(TP, L[21], L[22], AGP, NPn, 128, 128, 0);
        hgt_combine<<<(NPn + 7) / 8, 256, 0, stream>>>(HP, AGP, L[25], L[27], L[28], NPn);

        gelu_ew<<<((int)MszT + 255) / 256, 256, 0, stream>>>(AGT, TT2, (int)MszT);
        gemm(TT2, L[23], L[24], AGT, NTn, 128, 128, 0);
        hgt_combine<<<(NTn + 7) / 8, 256, 0, stream>>>(HT, AGT, L[26], L[29], L[30], NTn);
    }

    // ---- TopK pooling + outputs -------------------------------------------
    wnorm_k<<<1, 128, 0, stream>>>(P[90], SC1);
    score_k<<<(NPn + 7) / 8, 256, 0, stream>>>(HP, P[90], SC1, SCO, NPn);
    topk_pool<<<Bg, 256, 0, stream>>>(HP, SCO, (float*)d_out);
    copy_f32<<<((int)MszT + 255) / 256, 256, 0, stream>>>(
        HT, (float*)d_out + (size_t)Bg * 128, (int)MszT);
}
